// MultiHeadedAttention_19404662243359
// MI455X (gfx1250) — matrix operand order repaired and checked by reading, not compiled, not run on hardware
//
#include <hip/hip_runtime.h>

#define BB 2
#define SS 2048
#define DD 1024
#define HH 16
#define DHH 64

typedef __attribute__((ext_vector_type(16))) __bf16 v16bf;
typedef __attribute__((ext_vector_type(8)))  float   v8f;
typedef __attribute__((ext_vector_type(2)))  __bf16  bf16x2;

#if __has_builtin(__builtin_amdgcn_tensor_load_to_lds) && \
    __has_builtin(__builtin_amdgcn_s_wait_tensorcnt)
#define HAVE_TDM 1
#else
#define HAVE_TDM 0
#endif

static __device__ __forceinline__ __bf16 f2bf(float f) { return (__bf16)f; }

static __device__ __forceinline__ v8f wmma_bf16(v16bf a, v16bf b, v8f c) {
  // D = A(16x32) * B(32x16) + C, f32 accumulate
  return __builtin_amdgcn_wmma_f32_16x16x32_bf16(false, a, false, b, (short)0, c,
                                                 false, false);
}

#if HAVE_TDM
typedef unsigned int u32x4 __attribute__((ext_vector_type(4)));
typedef int          i32x4 __attribute__((ext_vector_type(4)));
typedef int          i32x8 __attribute__((ext_vector_type(8)));

typedef __attribute__((address_space(3))) char lds_char_t;
static __device__ __forceinline__ unsigned lds_off(void* p) {
  return (unsigned)(unsigned long long)(lds_char_t*)p;
}

// 2-D TDM load: bf16 tile (tile_rows x tile_cols) from row-major tensor
// (row stride `row_stride` elems) at gaddr into contiguous LDS at lds_byte_off.
// D# bitfields per CDNA5 ISA (count=1, type=2,
// data_size=1 (2 bytes), no pad / iterate / gather / multicast).
// This toolchain exposes the 6-argument builtin (extra int32x8 group).
static __device__ __forceinline__ void tdm_load_2d_bf16(
    unsigned lds_byte_off, const void* gaddr, int tile_cols, int tile_rows,
    int tensor_cols, int tensor_rows, int row_stride) {
  const unsigned long long ga = (unsigned long long)gaddr;
  u32x4 g0;
  g0[0] = 1u;                                          // count=1 (valid, user)
  g0[1] = lds_byte_off;                                // lds_addr
  g0[2] = (unsigned)(ga & 0xffffffffu);                // global_addr[31:0]
  g0[3] = (unsigned)((ga >> 32) & 0x1ffffffu)          // global_addr[56:32]
          | (2u << 30);                                // type=2 ("image")
  i32x8 g1;
  g1[0] = 1 << 16;                                     // data_size=1 -> 2 bytes
  g1[1] = (int)(((unsigned)tensor_cols & 0xffffu) << 16);        // dim0[15:0]
  g1[2] = (int)((((unsigned)tensor_cols >> 16) & 0xffffu) |
                (((unsigned)tensor_rows & 0xffffu) << 16));      // dim0 hi | dim1 lo
  g1[3] = (int)((((unsigned)tensor_rows >> 16) & 0xffffu) |
                (((unsigned)tile_cols & 0xffffu) << 16));        // dim1 hi | tile_dim0
  g1[4] = (int)((unsigned)tile_rows & 0xffffu);        // tile_dim1 (tile_dim2=0)
  g1[5] = row_stride;                                  // tensor_dim0_stride[31:0]
  g1[6] = 0;                                           // stride hi | dim1_stride lo
  g1[7] = 0;
  const i32x4 z4 = {0, 0, 0, 0};
  const i32x8 z8 = {0, 0, 0, 0, 0, 0, 0, 0};
  __builtin_amdgcn_tensor_load_to_lds(g0, g1, z4, z4, z8, 0);
}
#endif  // HAVE_TDM

// ---- fragment loaders -------------------------------------------------------
// A fragment 16x32 (MxK): element (m,k) at base[m*ld + k]; bf16, k-pairs contiguous.
static __device__ __forceinline__ v16bf load_a_bf16(const __bf16* base, int ld) {
  const int lane = threadIdx.x & 31;
  const int m  = lane & 15;
  const int kg = (lane >> 4) << 4;                 // 0 or 16
  const __bf16* row = base + (size_t)m * ld;
  v16bf a;
#pragma unroll
  for (int p = 0; p < 8; ++p) {
    const int k = kg + 2 * p;
    const unsigned int u = *(const unsigned int*)(row + k);
    const bf16x2 t = __builtin_bit_cast(bf16x2, u);
    a[2 * p]     = t.x;
    a[2 * p + 1] = t.y;
  }
  return a;
}

static __device__ __forceinline__ v16bf load_a_f32(const float* base, int ld) {
  const int lane = threadIdx.x & 31;
  const int m  = lane & 15;
  const int kg = (lane >> 4) << 4;
  const float* row = base + (size_t)m * ld;
  v16bf a;
#pragma unroll
  for (int p = 0; p < 8; ++p) {
    const int k = kg + 2 * p;
    const float2 u = *(const float2*)(row + k);
    a[2 * p]     = f2bf(u.x);
    a[2 * p + 1] = f2bf(u.y);
  }
  return a;
}

// B fragment 32x16 (KxN): element (k,n) at base[n*ld + k]; k contiguous per n.
static __device__ __forceinline__ v16bf load_b_bf16(const __bf16* base, int ld) {
  const int lane = threadIdx.x & 31;
  const int n  = lane & 15;
  const int k0 = (lane >> 4) << 4;                 // 0 or 16
  const __bf16* col = base + (size_t)n * ld + k0;
  v16bf b;
#pragma unroll
  for (int p = 0; p < 8; ++p) {
    const unsigned int u = *(const unsigned int*)(col + 2 * p);
    const bf16x2 t = __builtin_bit_cast(bf16x2, u);
    b[2 * p]     = t.x;
    b[2 * p + 1] = t.y;
  }
  return b;
}

// B fragment 32x16 (KxN) from k-major (row-major V) storage: (k,n) at base[k*ld + n].
static __device__ __forceinline__ v16bf load_b_bf16_kmajor(const __bf16* base, int ld) {
  const int lane = threadIdx.x & 31;
  const int n  = lane & 15;
  const int k0 = (lane >> 4) << 4;
  v16bf b;
#pragma unroll
  for (int p = 0; p < 8; ++p) {
    b[2 * p]     = base[(size_t)(k0 + 2 * p)     * ld + n];
    b[2 * p + 1] = base[(size_t)(k0 + 2 * p + 1) * ld + n];
  }
  return b;
}

static __device__ __forceinline__ v16bf load_b_f32(const float* base, int ld) {
  const int lane = threadIdx.x & 31;
  const int n  = lane & 15;
  const int k0 = (lane >> 4) << 4;
  const float* col = base + (size_t)n * ld + k0;
  v16bf b;
#pragma unroll
  for (int p = 0; p < 8; ++p) {
    const float2 u = *(const float2*)(col + 2 * p);
    b[2 * p]     = f2bf(u.x);
    b[2 * p + 1] = f2bf(u.y);
  }
  return b;
}

// ---- kernel 1: fused QKV projection (x @ W.T + b), bf16 WMMA, f32 accum -----
__global__ __launch_bounds__(256) void qkv_proj_kernel(
    const float* __restrict__ q_in, const float* __restrict__ k_in,
    const float* __restrict__ v_in,
    const float* __restrict__ Wq, const float* __restrict__ bq,
    const float* __restrict__ Wk, const float* __restrict__ bk,
    const float* __restrict__ Wv, const float* __restrict__ bv,
    __bf16* __restrict__ Qb, __bf16* __restrict__ Kb, __bf16* __restrict__ Vb) {
  const int which = blockIdx.z;  // 0=Q 1=K 2=V
  const float* X; const float* W; const float* bias; __bf16* out; float scale;
  if (which == 0)      { X = q_in; W = Wq; bias = bq; out = Qb; scale = 0.125f; }
  else if (which == 1) { X = k_in; W = Wk; bias = bk; out = Kb; scale = 1.0f;   }
  else                 { X = v_in; W = Wv; bias = bv; out = Vb; scale = 1.0f;   }

  const int wave = threadIdx.x >> 5;
  const int lane = threadIdx.x & 31;
  const int m0 = blockIdx.y * 16;                // row tile in (B*S)
  const int n0 = blockIdx.x * 128 + wave * 16;   // col tile in D

  v8f acc = {};
#pragma unroll 1
  for (int kk = 0; kk < DD; kk += 32) {
    const v16bf a  = load_a_f32(X + (size_t)m0 * DD + kk, DD);
    const v16bf bf = load_b_f32(W + (size_t)n0 * DD + kk, DD);  // W[n][k]
    acc = wmma_bf16(a, bf, acc);
  }

  const int n   = n0 + (lane & 15);
  const int hi8 = (lane >> 4) * 8;
  const int h   = n >> 6;
  const int dh  = n & 63;
  const float bn = bias[n];
#pragma unroll
  for (int r = 0; r < 8; ++r) {
    const int m = m0 + r + hi8;            // row in B*S (tile never straddles b)
    const int b = m / SS;
    const int s = m % SS;
    const float v = (acc[r] + bn) * scale;
    out[(((size_t)b * HH + h) * SS + s) * DHH + dh] = f2bf(v);
  }
}

// ---- kernel 2: flash attention per (b, h, 16-row q tile) --------------------
// Double-buffered K/V staging: the TDM streams tile j+1 while all 8 waves
// compute on tile j (one s_wait_tensorcnt + one barrier per step).
__global__ __launch_bounds__(256) void attn_kernel(
    const __bf16* __restrict__ Qb, const __bf16* __restrict__ Kb,
    const __bf16* __restrict__ Vb, const unsigned char* __restrict__ mask,
    __bf16* __restrict__ Ctx, float* __restrict__ top_attn) {
  __shared__ __align__(16) __bf16 Ktile[2][32][DHH];  // ping-pong  8 KB
  __shared__ __align__(16) __bf16 Vtile[2][32][DHH];  // ping-pong  8 KB
  __shared__ __align__(16) __bf16 Pbuf[8][16][32];    // per-wave P 8 KB

  const int b    = blockIdx.z;
  const int h    = blockIdx.y;
  const int wave = threadIdx.x >> 5;
  const int lane = threadIdx.x & 31;
  const int q0   = (blockIdx.x * 8 + wave) * 16;

  const size_t headoff = ((size_t)b * HH + h) * SS * DHH;
  const __bf16* Qh = Qb + headoff;
  const __bf16* Kh = Kb + headoff;
  const __bf16* Vh = Vb + headoff;
  const unsigned char* mrow = mask + (size_t)b * SS * SS;

  // query fragments (dh chunks 0..31 / 32..63) held in registers
  const v16bf aq0 = load_a_bf16(Qh + (size_t)q0 * DHH + 0,  DHH);
  const v16bf aq1 = load_a_bf16(Qh + (size_t)q0 * DHH + 32, DHH);

  v8f acc[4] = {};
  float mrun[8], lrun[8];
#pragma unroll
  for (int r = 0; r < 8; ++r) { mrun[r] = -3.0e38f; lrun[r] = 0.0f; }

  const int nlo = lane & 15;
  const int hi8 = (lane >> 4) * 8;

  const int NSTEPS = SS / 32;

#if HAVE_TDM
  if (wave == 0) {   // prologue: post step-0 DMA into buffer 0
    tdm_load_2d_bf16(lds_off(&Ktile[0][0][0]), Kh, DHH, 32, DHH, SS, DHH);
    tdm_load_2d_bf16(lds_off(&Vtile[0][0][0]), Vh, DHH, 32, DHH, SS, DHH);
  }
#endif

#pragma unroll 1
  for (int js = 0; js < NSTEPS; ++js) {
    const int j   = js * 32;
    const int cur = js & 1;
#if HAVE_TDM
    if (wave == 0) __builtin_amdgcn_s_wait_tensorcnt(0);  // cur buffer landed
    __syncthreads();  // publish cur; orders step js-1 reads before next issue
    if (wave == 0 && js + 1 < NSTEPS) {   // post next step's DMA (other buffer)
      tdm_load_2d_bf16(lds_off(&Ktile[cur ^ 1][0][0]),
                       Kh + (size_t)(j + 32) * DHH, DHH, 32, DHH, SS, DHH);
      tdm_load_2d_bf16(lds_off(&Vtile[cur ^ 1][0][0]),
                       Vh + (size_t)(j + 32) * DHH, DHH, 32, DHH, SS, DHH);
    }
#else
    {   // fallback: manual contiguous 4 KB copies into cur buffer
      const uint4* ksrc = (const uint4*)(Kh + (size_t)j * DHH);
      const uint4* vsrc = (const uint4*)(Vh + (size_t)j * DHH);
      ((uint4*)&Ktile[cur][0][0])[threadIdx.x] = ksrc[threadIdx.x];
      ((uint4*)&Vtile[cur][0][0])[threadIdx.x] = vsrc[threadIdx.x];
      if (j + 32 < SS) {
        __builtin_prefetch(Kh + (size_t)(j + 32) * DHH, 0, 1);
        __builtin_prefetch(Vh + (size_t)(j + 32) * DHH, 0, 1);
      }
    }
    __syncthreads();
#endif

    // scores: two 16-key tiles, K-dim = DH split as 2x32
    v8f s0 = {}, s1 = {};
    s0 = wmma_bf16(aq0, load_b_bf16(&Ktile[cur][0][0],   DHH), s0);
    s0 = wmma_bf16(aq1, load_b_bf16(&Ktile[cur][0][32],  DHH), s0);
    s1 = wmma_bf16(aq0, load_b_bf16(&Ktile[cur][16][0],  DHH), s1);
    s1 = wmma_bf16(aq1, load_b_bf16(&Ktile[cur][16][32], DHH), s1);

    // mask (True -> -1e18) + online softmax update
#pragma unroll
    for (int r = 0; r < 8; ++r) {
      const int row = q0 + r + hi8;
      const unsigned char* mr = mrow + (size_t)row * SS + j;
      s0[r] = mr[nlo]      ? -1.0e18f : s0[r];
      s1[r] = mr[16 + nlo] ? -1.0e18f : s1[r];
    }
    float sc[8];
#pragma unroll
    for (int r = 0; r < 8; ++r) {
      float tm = fmaxf(s0[r], s1[r]);
#pragma unroll
      for (int off = 8; off >= 1; off >>= 1)
        tm = fmaxf(tm, __shfl_xor(tm, off, 16));     // reduce within half-wave
      const float mnew = fmaxf(mrun[r], tm);
      sc[r] = __expf(mrun[r] - mnew);
      mrun[r] = mnew;
      const float p0 = __expf(s0[r] - mnew);
      const float p1 = __expf(s1[r] - mnew);
      s0[r] = p0; s1[r] = p1;
      float rs = p0 + p1;
#pragma unroll
      for (int off = 8; off >= 1; off >>= 1)
        rs += __shfl_xor(rs, off, 16);
      lrun[r] = lrun[r] * sc[r] + rs;
    }
#pragma unroll
    for (int nf = 0; nf < 4; ++nf)
#pragma unroll
      for (int r = 0; r < 8; ++r) acc[nf][r] *= sc[r];

    // C-layout probs -> A-layout bf16 via wave-private LDS (same-wave DS order)
#pragma unroll
    for (int r = 0; r < 8; ++r) {
      const int m = r + hi8;
      Pbuf[wave][m][nlo]      = f2bf(s0[r]);
      Pbuf[wave][m][16 + nlo] = f2bf(s1[r]);
    }
    const v16bf ap = load_a_bf16(&Pbuf[wave][0][0], 32);
#pragma unroll
    for (int nf = 0; nf < 4; ++nf)
      acc[nf] = wmma_bf16(ap, load_b_bf16_kmajor(&Vtile[cur][0][nf * 16], DHH),
                          acc[nf]);
  }

  // normalize + store ctx (bf16, (B,S,D) row-major)
  float inv[8];
#pragma unroll
  for (int r = 0; r < 8; ++r) inv[r] = 1.0f / lrun[r];
#pragma unroll
  for (int nf = 0; nf < 4; ++nf) {
    const int d = h * DHH + nf * 16 + nlo;
#pragma unroll
    for (int r = 0; r < 8; ++r) {
      const int srow = q0 + r + hi8;
      Ctx[((size_t)b * SS + srow) * DD + d] = f2bf(acc[nf][r] * inv[r]);
    }
  }

  // pass 2 (head 0 only; uniform per block): emit normalized attention probs
  if (h == 0) {
#pragma unroll 1
    for (int j = 0; j < SS; j += 16) {
      v8f s = {};
      s = wmma_bf16(aq0, load_b_bf16(Kh + (size_t)j * DHH + 0,  DHH), s);
      s = wmma_bf16(aq1, load_b_bf16(Kh + (size_t)j * DHH + 32, DHH), s);
#pragma unroll
      for (int r = 0; r < 8; ++r) {
        const int row = q0 + r + hi8;
        const int col = j + nlo;
        float v = mrow[(size_t)row * SS + col] ? -1.0e18f : s[r];
        top_attn[((size_t)b * SS + row) * SS + col] = __expf(v - mrun[r]) * inv[r];
      }
    }
  }
}

// ---- kernel 3: output projection ctx @ Wo.T + bo ----------------------------
__global__ __launch_bounds__(256) void out_proj_kernel(
    const __bf16* __restrict__ Ctx, const float* __restrict__ Wo,
    const float* __restrict__ bo, float* __restrict__ out) {
  const int wave = threadIdx.x >> 5;
  const int lane = threadIdx.x & 31;
  const int m0 = blockIdx.y * 16;
  const int n0 = blockIdx.x * 128 + wave * 16;

  v8f acc = {};
#pragma unroll 1
  for (int kk = 0; kk < DD; kk += 32) {
    const v16bf a  = load_a_bf16(Ctx + (size_t)m0 * DD + kk, DD);
    const v16bf bf = load_b_f32(Wo + (size_t)n0 * DD + kk, DD);
    acc = wmma_bf16(a, bf, acc);
  }
  const int n   = n0 + (lane & 15);
  const int hi8 = (lane >> 4) * 8;
  const float bn = bo[n];
#pragma unroll
  for (int r = 0; r < 8; ++r) {
    const int m = m0 + r + hi8;
    out[(size_t)m * DD + n] = acc[r] + bn;
  }
}

extern "C" void kernel_launch(void* const* d_in, const int* in_sizes, int n_in,
                              void* d_out, int out_size, void* d_ws, size_t ws_size,
                              hipStream_t stream) {
  (void)in_sizes; (void)n_in; (void)out_size; (void)ws_size;
  const float* key   = (const float*)d_in[0];
  const float* value = (const float*)d_in[1];
  const float* query = (const float*)d_in[2];
  const unsigned char* mask = (const unsigned char*)d_in[3];
  const float* Wk = (const float*)d_in[4];  const float* bk = (const float*)d_in[5];
  const float* Wv = (const float*)d_in[6];  const float* bv = (const float*)d_in[7];
  const float* Wq = (const float*)d_in[8];  const float* bq = (const float*)d_in[9];
  const float* Wo = (const float*)d_in[10]; const float* bo = (const float*)d_in[11];

  float* out = (float*)d_out;
  float* top = out + (size_t)BB * SS * DD;

  const size_t nElem = (size_t)BB * SS * DD;   // 4 Mi elements
  __bf16* Qb  = (__bf16*)d_ws;                 // bf16 workspace: 4 x 8 MB = 32 MB
  __bf16* Kb  = Qb + nElem;
  __bf16* Vb  = Kb + nElem;
  __bf16* Ctx = Vb + nElem;

  const dim3 blk(256);
  qkv_proj_kernel<<<dim3(DD / 128, (BB * SS) / 16, 3), blk, 0, stream>>>(
      query, key, value, Wq, bq, Wk, bk, Wv, bv, Qb, Kb, Vb);
  attn_kernel<<<dim3(SS / 16 / 8, HH, BB), blk, 0, stream>>>(
      Qb, Kb, Vb, mask, Ctx, top);
  out_proj_kernel<<<dim3(DD / 128, (BB * SS) / 16, 1), blk, 0, stream>>>(
      Ctx, Wo, bo, out);
}
